// AttentiveNCDE_78383153152568
// MI455X (gfx1250) — compile-verified, matrix operand order checked by reading, not run on hardware
//
#include <hip/hip_runtime.h>
#include <hip/hip_bf16.h>

// ---------------------------------------------------------------------------
// Problem constants (from reference): T=100, B=1024, I=256, H=256, O=128
// ---------------------------------------------------------------------------
#define TT  100
#define BB  1024
#define II  256
#define HH  256
#define OO  128
#define H3  768

// bf16 weight layout inside d_ws (ushort element offsets)
#define OFF_WIH 0
#define OFF_WHH (OFF_WIH + H3*II)        // 196608
#define OFF_FW1 (OFF_WHH + H3*HH)        // 393216
#define OFF_FW2 (OFF_FW1 + HH*HH)        // 458752
#define OFF_OW  (OFF_FW2 + HH*HH)        // 524288
#define W_TOTAL (OFF_OW  + OO*HH)        // 557056 ushorts

// LDS strides (padded: bf16 +8 halves, f32 +4 dwords -> bank-conflict friendly)
#define SB 264   // bf16 buffers of logical width 256
#define SF 260   // f32  buffers of logical width 256
#define SG 772   // f32  buffers of logical width 768

// LDS carve sizes (bytes)
#define SMEM_BYTES (4*16*SB*2 /*hb,xb,vb,ub*/ + 3*16*SF*4 /*hf,acc,tb*/ + 2*16*SG*4 /*gi,gh*/)

typedef __attribute__((ext_vector_type(8)))  short   short8;
typedef __attribute__((ext_vector_type(16))) short   short16;
typedef __attribute__((ext_vector_type(16))) __bf16  v16bf;
typedef __attribute__((ext_vector_type(8)))  float   v8f;

// float -> bf16 (round to nearest even)
__device__ __forceinline__ unsigned short f2bf(float f) {
    unsigned u = __float_as_uint(f);
    u += 0x7FFFu + ((u >> 16) & 1u);
    return (unsigned short)(u >> 16);
}

__device__ __forceinline__ float fast_sigmoid(float x) {
    return __fdividef(1.0f, 1.0f + __expf(-x));
}
__device__ __forceinline__ float fast_tanh(float x) {
    float e = __expf(-2.0f * x);
    return __fdividef(1.0f - e, 1.0f + e);
}

// Load one 16x32 bf16 fragment (A-layout; B uses the mirrored layout with
// row = N from [N][K] row-major weights == column-major B matrix).
// Per lane: 16 bf16 = two contiguous 8-element (16B) runs: [kb, kb+8) and
// [kb+16, kb+24), kb = k0 + (lane>=16 ? 8 : 0).
__device__ __forceinline__ v16bf load_frag(const unsigned short* base,
                                           int rowStride, int row, int kb) {
    const unsigned short* p = base + row * rowStride + kb;
    short8 lo = *(const short8*)(p);
    short8 hi = *(const short8*)(p + 16);
    short16 s = __builtin_shufflevector(lo, hi, 0,1,2,3,4,5,6,7,8,9,10,11,12,13,14,15);
    return __builtin_bit_cast(v16bf, s);
}

__device__ __forceinline__ v8f wmma_bf16(v16bf a, v16bf b, v8f c) {
    return __builtin_amdgcn_wmma_f32_16x16x32_bf16(false, a, false, b,
                                                   (short)0, c, false, false);
}

// Store a 16x16 f32 C/D fragment into an LDS f32 tile.
// C/D layout: VGPR r -> M = r + 8*(lane>=16), N = lane&15.
__device__ __forceinline__ void store_tile(float* base, int rowStride, int n0,
                                           int lane, v8f c) {
    const int n  = lane & 15;
    const int mh = (lane >> 4) * 8;
#pragma unroll
    for (int r = 0; r < 8; ++r)
        base[(mh + r) * rowStride + n0 + n] = c[r];
}

// ---------------------------------------------------------------------------
// Prep: convert all weights to bf16, row-major [N][K], into d_ws
// ---------------------------------------------------------------------------
__global__ void ncde_prep(const float* __restrict__ wih, const float* __restrict__ whh,
                          const float* __restrict__ fw1, const float* __restrict__ fw2,
                          const float* __restrict__ ow,  unsigned short* __restrict__ dst) {
    int i = blockIdx.x * blockDim.x + threadIdx.x;
    float v;
    if      (i < OFF_WHH) v = wih[i - OFF_WIH];
    else if (i < OFF_FW1) v = whh[i - OFF_WHH];
    else if (i < OFF_FW2) v = fw1[i - OFF_FW1];
    else if (i < OFF_OW)  v = fw2[i - OFF_FW2];
    else if (i < W_TOTAL) v = ow [i - OFF_OW];
    else return;
    dst[i] = f2bf(v);
}

// ---------------------------------------------------------------------------
// Main persistent kernel: one workgroup owns a 16-row batch tile for all T.
// 256 threads = 8 wave32s.
// ---------------------------------------------------------------------------
__global__ void __launch_bounds__(256, 1)
ncde_main(const float* __restrict__ timep, const float* __restrict__ h0,
          const float* __restrict__ xin,
          const float* __restrict__ b_ih, const float* __restrict__ b_hh,
          const float* __restrict__ f_b1, const float* __restrict__ f_b2,
          const float* __restrict__ out_b,
          const unsigned short* __restrict__ wq, float* __restrict__ out) {
    extern __shared__ char smem[];
    unsigned short* hb = (unsigned short*)smem;           // bf16(h)      16 x SB
    unsigned short* xb = hb + 16 * SB;                    // bf16(x_t)    16 x SB
    unsigned short* vb = xb + 16 * SB;                    // RK4 stage in 16 x SB
    unsigned short* ub = vb + 16 * SB;                    // relu hidden  16 x SB
    float* hf  = (float*)(ub + 16 * SB);                  // h (f32)      16 x SF
    float* acc = hf  + 16 * SF;                           // RK4 accum    16 x SF
    float* tb  = acc + 16 * SF;                           // GEMM scratch 16 x SF
    float* gi  = tb  + 16 * SF;                           // x-gates      16 x SG
    float* gh  = gi  + 16 * SG;                           // h-gates      16 x SG

    const unsigned short* wih = wq + OFF_WIH;
    const unsigned short* whh = wq + OFF_WHH;
    const unsigned short* fw1 = wq + OFF_FW1;
    const unsigned short* fw2 = wq + OFF_FW2;
    const unsigned short* ow  = wq + OFF_OW;

    const int tid  = threadIdx.x;        // 0..255 ; column j = tid
    const int lane = tid & 31;
    const int wave = tid >> 5;           // 0..7
    const int m0   = blockIdx.x * 16;    // batch-row tile base
    const int mrow = lane & 15;
    const int koff = (lane & 16) ? 8 : 0;

    // per-thread biases (column j = tid is fixed for elementwise phases)
    const float bir = b_ih[tid], biz = b_ih[HH + tid], bin = b_ih[2 * HH + tid];
    const float bhr = b_hh[tid], bhz = b_hh[HH + tid], bhn = b_hh[2 * HH + tid];
    const float fb1 = f_b1[tid], fb2 = f_b2[tid];

    // init state
#pragma unroll 4
    for (int it = 0; it < 16; ++it) {
        float v = h0[(size_t)(m0 + it) * HH + tid];
        hf[it * SF + tid] = v;
        hb[it * SB + tid] = f2bf(v);
    }
    __syncthreads();

    for (int t = 0; t < TT - 1; ++t) {
        const float dt = timep[t + 1] - timep[t];

        // prefetch next step's x tile (contiguous 16KB): 256 threads x 64B
        if (t + 1 < TT - 1) {
            const char* nx =
                (const char*)&xin[((size_t)(t + 1) * BB + m0) * II];
            __builtin_prefetch(nx + tid * 64, 0, 3);
        }

        // stage x_t tile as bf16
#pragma unroll 4
        for (int it = 0; it < 16; ++it)
            xb[it * SB + tid] = f2bf(xin[((size_t)t * BB + m0 + it) * II + tid]);
        __syncthreads();

        // ---- GRU gate GEMMs: gi = x @ w_ih^T ; gh = h @ w_hh^T  (N=768) ----
        {
            // hoist A fragments (shared by all 6 N-tiles of this wave)
            v16bf axr[8], ahr[8];
#pragma unroll
            for (int kk = 0; kk < 8; ++kk) {
                axr[kk] = load_frag(xb, SB, mrow, kk * 32 + koff);
                ahr[kk] = load_frag(hb, SB, mrow, kk * 32 + koff);
            }
#pragma unroll 1
            for (int q = 0; q < 6; ++q) {
                const int n0 = (wave + 8 * q) * 16;
                v8f cg = {}; v8f ch = {};
#pragma unroll
                for (int kk = 0; kk < 8; ++kk) {
                    const int kb = kk * 32 + koff;
                    v16bf bi = load_frag(wih, II, n0 + mrow, kb);
                    v16bf bh = load_frag(whh, HH, n0 + mrow, kb);
                    cg = wmma_bf16(axr[kk], bi, cg);
                    ch = wmma_bf16(ahr[kk], bh, ch);
                }
                store_tile(gi, SG, n0, lane, cg);
                store_tile(gh, SG, n0, lane, ch);
            }
        }
        __syncthreads();

        // ---- GRU elementwise (attention gate in reference is dead code) ----
#pragma unroll 4
        for (int it = 0; it < 16; ++it) {
            float gr = gi[it * SG + tid]          + bir + gh[it * SG + tid]          + bhr;
            float gz = gi[it * SG + HH + tid]     + biz + gh[it * SG + HH + tid]     + bhz;
            float gn = gi[it * SG + 2 * HH + tid] + bin;
            float hn = gh[it * SG + 2 * HH + tid] + bhn;
            float r = fast_sigmoid(gr);
            float z = fast_sigmoid(gz);
            float n = fast_tanh(gn + r * hn);
            float hv = hf[it * SF + tid];
            hv = (1.0f - z) * n + z * hv;
            hf[it * SF + tid]  = hv;
            unsigned short hq  = f2bf(hv);
            hb[it * SB + tid]  = hq;
            vb[it * SB + tid]  = hq;        // RK4 stage-0 input
            acc[it * SF + tid] = 0.0f;
        }
        __syncthreads();

        // ---- RK4 on f(v) = relu(v @ W1^T + b1) @ W2^T + b2 ----
        {
            // cache this wave's W1/W2 fragments in registers for all 4 stages
            v16bf w1r[2][8], w2r[2][8];
#pragma unroll
            for (int q = 0; q < 2; ++q) {
                const int n0 = (wave + 8 * q) * 16;
#pragma unroll
                for (int kk = 0; kk < 8; ++kk) {
                    const int kb = kk * 32 + koff;
                    w1r[q][kk] = load_frag(fw1, HH, n0 + mrow, kb);
                    w2r[q][kk] = load_frag(fw2, HH, n0 + mrow, kb);
                }
            }

#pragma unroll 1
            for (int s = 0; s < 4; ++s) {
                const float cs = (s == 2) ? 1.0f : 0.5f;             // used for s<3
                const float wk = (s == 1 || s == 2) ? 2.0f : 1.0f;

                // tb = vb @ W1^T  (A-frags hoisted, B from registers)
                {
                    v16bf af[8];
#pragma unroll
                    for (int kk = 0; kk < 8; ++kk)
                        af[kk] = load_frag(vb, SB, mrow, kk * 32 + koff);
#pragma unroll
                    for (int q = 0; q < 2; ++q) {
                        v8f c = {};
#pragma unroll
                        for (int kk = 0; kk < 8; ++kk)
                            c = wmma_bf16(af[kk], w1r[q][kk], c);
                        store_tile(tb, SF, (wave + 8 * q) * 16, lane, c);
                    }
                }
                __syncthreads();

                // ub = bf16(relu(tb + b1))
#pragma unroll 4
                for (int it = 0; it < 16; ++it) {
                    float u = tb[it * SF + tid] + fb1;
                    ub[it * SB + tid] = f2bf(u > 0.0f ? u : 0.0f);
                }
                __syncthreads();

                // tb = ub @ W2^T
                {
                    v16bf af[8];
#pragma unroll
                    for (int kk = 0; kk < 8; ++kk)
                        af[kk] = load_frag(ub, SB, mrow, kk * 32 + koff);
#pragma unroll
                    for (int q = 0; q < 2; ++q) {
                        v8f c = {};
#pragma unroll
                        for (int kk = 0; kk < 8; ++kk)
                            c = wmma_bf16(af[kk], w2r[q][kk], c);
                        store_tile(tb, SF, (wave + 8 * q) * 16, lane, c);
                    }
                }
                __syncthreads();

                // k_s = tb + b2 ; accumulate ; prepare next stage input
#pragma unroll 4
                for (int it = 0; it < 16; ++it) {
                    float k = tb[it * SF + tid] + fb2;
                    acc[it * SF + tid] += wk * k;
                    if (s < 3)
                        vb[it * SB + tid] = f2bf(hf[it * SF + tid] + cs * dt * k);
                }
                __syncthreads();
            }
        }

        // h += (dt/6) * (k1 + 2k2 + 2k3 + k4)
#pragma unroll 4
        for (int it = 0; it < 16; ++it) {
            float hv = hf[it * SF + tid] + (dt * (1.0f / 6.0f)) * acc[it * SF + tid];
            hf[it * SF + tid] = hv;
            hb[it * SB + tid] = f2bf(hv);
        }
        __syncthreads();
    }

    // ---- output projection: out = h @ out_w^T + out_b  (N=128, 1 tile/wave) ----
    {
        const int n0 = wave * 16;
        v8f c = {};
#pragma unroll
        for (int k0 = 0; k0 < HH; k0 += 32) {
            const int kb = k0 + koff;
            v16bf a = load_frag(hb, SB, mrow, kb);
            v16bf b = load_frag(ow, HH, n0 + mrow, kb);
            c = wmma_bf16(a, b, c);
        }
        const int n  = lane & 15;
        const int mh = (lane >> 4) * 8;
        const float bo = out_b[n0 + n];
#pragma unroll
        for (int r = 0; r < 8; ++r)
            out[(size_t)(m0 + mh + r) * OO + n0 + n] = c[r] + bo;
    }
}

// ---------------------------------------------------------------------------
extern "C" void kernel_launch(void* const* d_in, const int* in_sizes, int n_in,
                              void* d_out, int out_size, void* d_ws, size_t ws_size,
                              hipStream_t stream) {
    const float* timep = (const float*)d_in[0];
    const float* h0    = (const float*)d_in[1];
    const float* xin   = (const float*)d_in[2];
    const float* w_ih  = (const float*)d_in[3];
    const float* w_hh  = (const float*)d_in[4];
    const float* b_ih  = (const float*)d_in[5];
    const float* b_hh  = (const float*)d_in[6];
    // d_in[7], d_in[8] = attn_w / attn_b : dead code in the reference
    const float* f_w1  = (const float*)d_in[9];
    const float* f_b1  = (const float*)d_in[10];
    const float* f_w2  = (const float*)d_in[11];
    const float* f_b2  = (const float*)d_in[12];
    const float* out_w = (const float*)d_in[13];
    const float* out_b = (const float*)d_in[14];

    unsigned short* wq = (unsigned short*)d_ws;   // ~1.1 MB of bf16 weights

    ncde_prep<<<(W_TOTAL + 255) / 256, 256, 0, stream>>>(w_ih, w_hh, f_w1, f_w2,
                                                         out_w, wq);

    hipFuncSetAttribute(reinterpret_cast<const void*>(ncde_main),
                        hipFuncAttributeMaxDynamicSharedMemorySize, SMEM_BYTES);

    ncde_main<<<BB / 16, 256, SMEM_BYTES, stream>>>(timep, h0, xin, b_ih, b_hh,
                                                    f_b1, f_b2, out_b, wq,
                                                    (float*)d_out);
}
